// MultiHeadAttention_24567212933961
// MI455X (gfx1250) — matrix operand order repaired and checked by reading, not compiled, not run on hardware
//
#include <hip/hip_runtime.h>
#include <hip/hip_bf16.h>

typedef _Float16 half_t;
typedef __attribute__((ext_vector_type(16))) _Float16 v16h;
typedef __attribute__((ext_vector_type(8)))  _Float16 v8h;
typedef __attribute__((ext_vector_type(4)))  _Float16 v4h;
typedef __attribute__((ext_vector_type(8)))  float    v8f;
typedef __attribute__((ext_vector_type(4)))  unsigned v4u;
typedef __attribute__((ext_vector_type(8)))  unsigned v8u;

#define D_MODEL   2048
#define NUM_HEADS 16
#define HEAD_DIM  128
#define SEQ       2048
#define BATCH     2
#define MTOT      (BATCH * SEQ)             // 4096
#define QK_SCALE  0.08838834764831845f      // 1/sqrt(128)

// ---------------------------------------------------------------- helpers

__device__ __forceinline__ v8f zero_v8f() {
  v8f z;
#pragma unroll
  for (int i = 0; i < 8; ++i) z[i] = 0.f;
  return z;
}

__device__ __forceinline__ v16h make_v16(v8h lo, v8h hi) {
  v16h r;
#pragma unroll
  for (int i = 0; i < 8; ++i) { r[i] = lo[i]; r[i + 8] = hi[i]; }
  return r;
}

// A fragment, 16(M) x 32(K) f16. `tile` points at (m=0,k=0); stride in halves.
// Lane L: row m = L%16; halves k = g*8..g*8+7 and 16+g*8..+7 (g = L/16).
__device__ __forceinline__ v16h load_a_frag(const half_t* tile, int stride, int lane) {
  const int m = lane & 15, g = lane >> 4;
  const half_t* p = tile + m * stride + g * 8;
  v8h lo = *(const v8h*)(p);
  v8h hi = *(const v8h*)(p + 16);
  return make_v16(lo, hi);
}

// B fragment, 32(K) x 16(N) f16, with B stored "n-major" (row n contiguous in k).
// Lane L: col n = L%16; halves k = g*8..g*8+7 and 16+g*8..+7 (g = L/16).
__device__ __forceinline__ v16h load_b_frag(const half_t* tile, int stride, int lane) {
  const int n = lane & 15, g = lane >> 4;
  const half_t* p = tile + n * stride + g * 8;
  v8h lo = *(const v8h*)(p);
  v8h hi = *(const v8h*)(p + 16);
  return make_v16(lo, hi);
}

__device__ __forceinline__ v8f wmma_f16(v16h a, v16h b, v8f c) {
  return __builtin_amdgcn_wmma_f32_16x16x32_f16(
      /*neg_a=*/false, a, /*neg_b=*/false, b,
      /*c_mod=*/(short)0, c, /*reuse_a=*/false, /*reuse_b=*/false);
}

// CDNA5 async global->LDS copy (16B per lane), tracked by ASYNCcnt.
// GVS mode: 64-bit uniform SGPR base + 32-bit per-lane byte offset.
__device__ __forceinline__ void async_copy_b128(unsigned lds_byte_addr,
                                                unsigned long long sbase,
                                                unsigned voff_bytes) {
  asm volatile("global_load_async_to_lds_b128 %0, %1, %2"
               :: "v"(lds_byte_addr), "v"(voff_bytes), "s"(sbase)
               : "memory");
}

// CDNA5 Tensor Data Mover: 2D tile DMA global->LDS, tracked by TENSORcnt.
// Descriptor per CDNA5 ISA (groups 0 and 1; 2D => VADDR2/3 null).
// data_size=1 (2-byte elements), pad_enable=1 so the TDM inserts LDS padding
// (bank-conflict-free row strides) by itself.
__device__ __forceinline__ void tdm_load_2d(unsigned lds_addr, const void* gaddr,
                                            unsigned tile_d0, unsigned tile_d1,
                                            unsigned tensor_d0, unsigned tensor_d1,
                                            unsigned stride0,
                                            unsigned pad_interval_code,
                                            unsigned pad_amount_code) {
  const unsigned long long ga = (unsigned long long)gaddr;
  v4u g0;
  g0[0] = 1u;                                    // count=1, user descriptor
  g0[1] = lds_addr;                              // LDS byte address
  g0[2] = (unsigned)ga;                          // global_addr[31:0]
  g0[3] = (unsigned)(ga >> 32) | (2u << 30);     // global_addr[56:32] | type=2
  v8u g1;
  g1[0] = (1u << 16)                             // data_size = 2 bytes
        | (1u << 20)                             // pad_enable
        | (pad_interval_code << 22)
        | (pad_amount_code << 25);
  g1[1] = (tensor_d0 & 0xFFFFu) << 16;           // [47:32]=atomic_barrier=0, dim0 lo
  g1[2] = (tensor_d0 >> 16) | ((tensor_d1 & 0xFFFFu) << 16);
  g1[3] = (tensor_d1 >> 16) | (tile_d0 << 16);
  g1[4] = tile_d1;                               // tile_dim1; tile_dim2=0
  g1[5] = stride0;                               // tensor_dim0_stride[31:0]
  g1[6] = 0;                                     // stride0 hi | stride1 lo
  g1[7] = 0;                                     // stride1 hi
  asm volatile("tensor_load_to_lds %0, %1" :: "s"(g0), "s"(g1) : "memory");
}

// ---------------------------------------------------------------- convert

__global__ void cvt_f32_f16(const float* __restrict__ src,
                            half_t* __restrict__ dst, int n4) {
  int i = blockIdx.x * blockDim.x + threadIdx.x;
  if (i < n4) {
    float4 v = ((const float4*)src)[i];
    v4h o;
    o[0] = (half_t)v.x; o[1] = (half_t)v.y; o[2] = (half_t)v.z; o[3] = (half_t)v.w;
    ((v4h*)dst)[i] = o;
  }
}

// ---------------------------------------------------------------- GEMM
// C(M=4096, N=2048) = A(M,2048) @ W^T + bias, A/W in f16.
// Double-buffered LDS staging via GLOBAL_LOAD_ASYNC_TO_LDS_B128 + S_WAIT_ASYNCCNT.
// mode 0: Q -> f16 (B,H,S,Dh) scaled; 1: K -> f16 (B,H,S,Dh);
// mode 2: V -> f16 (B,H,Dh,S); 3: O -> f32 (M,N).

#define BK  32
#define LDT 40   // padded LDS stride (halves): 80B -> conflict-free b128 frags

__global__ __launch_bounds__(256)
void gemm_wmma(const half_t* __restrict__ A, const half_t* __restrict__ Bw,
               const float* __restrict__ bias,
               half_t* __restrict__ outH, float* __restrict__ outF, int mode) {
  __shared__ __attribute__((aligned(16))) half_t Ash[2][128 * LDT];
  __shared__ __attribute__((aligned(16))) half_t Bsh[2][128 * LDT];

  const int tid  = threadIdx.x;
  const int lane = tid & 31, w = tid >> 5;
  const int wm = (w & 3) * 32;
  const int wn = (w >> 2) * 64;
  const int m0 = blockIdx.y * 128, n0 = blockIdx.x * 128;
  const int row = tid >> 1, seg = (tid & 1) * 16;

  v8f acc[2][4];
#pragma unroll
  for (int i = 0; i < 2; ++i)
#pragma unroll
    for (int j = 0; j < 4; ++j) acc[i][j] = zero_v8f();

  const unsigned long long Abase = (unsigned long long)(A  + (size_t)m0 * D_MODEL);
  const unsigned long long Bbase = (unsigned long long)(Bw + (size_t)n0 * D_MODEL);
  const unsigned rowoff = (unsigned)(row * D_MODEL + seg) * 2u;

  unsigned ldsA[2], ldsB[2];
#pragma unroll
  for (int s = 0; s < 2; ++s) {
    ldsA[s] = (unsigned)(unsigned long long)(&Ash[s][row * LDT + seg]);
    ldsB[s] = (unsigned)(unsigned long long)(&Bsh[s][row * LDT + seg]);
  }

  {
    const unsigned go = rowoff;
    async_copy_b128(ldsA[0],      Abase, go);
    async_copy_b128(ldsA[0] + 16, Abase, go + 16);
    async_copy_b128(ldsB[0],      Bbase, go);
    async_copy_b128(ldsB[0] + 16, Bbase, go + 16);
  }

  int buf = 0;
  for (int kt = 0; kt < D_MODEL; kt += BK, buf ^= 1) {
    if (kt + BK < D_MODEL) {
      const unsigned go = rowoff + (unsigned)(kt + BK) * 2u;
      const int nb = buf ^ 1;
      async_copy_b128(ldsA[nb],      Abase, go);
      async_copy_b128(ldsA[nb] + 16, Abase, go + 16);
      async_copy_b128(ldsB[nb],      Bbase, go);
      async_copy_b128(ldsB[nb] + 16, Bbase, go + 16);
      asm volatile("s_wait_asynccnt 0x4" ::: "memory");
    } else {
      asm volatile("s_wait_asynccnt 0x0" ::: "memory");
    }
    __syncthreads();

    v16h af[2], bf[4];
#pragma unroll
    for (int mt = 0; mt < 2; ++mt)
      af[mt] = load_a_frag(&Ash[buf][(wm + mt * 16) * LDT], LDT, lane);
#pragma unroll
    for (int nt = 0; nt < 4; ++nt)
      bf[nt] = load_b_frag(&Bsh[buf][(wn + nt * 16) * LDT], LDT, lane);
#pragma unroll
    for (int mt = 0; mt < 2; ++mt)
#pragma unroll
      for (int nt = 0; nt < 4; ++nt)
        acc[mt][nt] = wmma_f16(af[mt], bf[nt], acc[mt][nt]);

    __syncthreads();
  }

  const int ln = lane & 15, g = lane >> 4;
#pragma unroll
  for (int nt = 0; nt < 4; ++nt) {
    const int n  = n0 + wn + nt * 16 + ln;
    const float bv = bias[n];
    const int h = n >> 7, dh = n & 127;
#pragma unroll
    for (int mt = 0; mt < 2; ++mt) {
#pragma unroll
      for (int r = 0; r < 8; ++r) {
        const int m = m0 + wm + mt * 16 + r + 8 * g;
        const float v = acc[mt][nt][r] + bv;
        const int b = m >> 11, s = m & 2047;
        if (mode == 0) {
          outH[((size_t)(b * NUM_HEADS + h) * SEQ + s) * HEAD_DIM + dh] =
              (half_t)(v * QK_SCALE);
        } else if (mode == 1) {
          outH[((size_t)(b * NUM_HEADS + h) * SEQ + s) * HEAD_DIM + dh] = (half_t)v;
        } else if (mode == 2) {
          outH[((size_t)(b * NUM_HEADS + h) * HEAD_DIM + dh) * SEQ + s] = (half_t)v;
        } else {
          outF[(size_t)m * D_MODEL + n] = v;
        }
      }
    }
  }
}

// ---------------------------------------------------------------- flash attention
// Grid (S/64, H, B), 128 threads = 4 waves. Each wave owns one 16-row query tile.
// K/V 32-key tiles are DMA'd into LDS by the Tensor Data Mover (wave 0 issues,
// TENSORcnt + barrier publishes), double-buffered one block ahead.
// TDM pad codes: K rows 128 halves = 64 DW -> interval code 5, +4 DW pad (code 3)
//   => LDS row stride 136 halves. V rows 32 halves = 16 DW -> interval code 3,
//   +4 DW pad (code 3) => LDS row stride 40 halves.

#define KLD 136
#define VLD 40
#define PLD 40

__global__ __launch_bounds__(128)
void flash_attn(const half_t* __restrict__ Qh, const half_t* __restrict__ Kh,
                const half_t* __restrict__ Vt, half_t* __restrict__ ctx) {
  __shared__ __attribute__((aligned(16))) half_t Ksh[2][32 * KLD];
  __shared__ __attribute__((aligned(16))) half_t Vsh[2][128 * VLD];
  __shared__ __attribute__((aligned(16))) half_t Pl[4][16 * PLD];

  const int lane = threadIdx.x & 31, w = threadIdx.x >> 5;
  const int h = blockIdx.y, b = blockIdx.z;
  const int bh = b * NUM_HEADS + h;
  const int qs = blockIdx.x * 64 + w * 16;
  const int ln = lane & 15, g = lane >> 4;

  const half_t* Khead = Kh + (size_t)bh * SEQ * HEAD_DIM;
  const half_t* Vhead = Vt + (size_t)bh * HEAD_DIM * SEQ;

  // resident Q fragments (already scaled by 1/sqrt(Dh) at projection time)
  const half_t* Qbase = Qh + ((size_t)bh * SEQ + qs) * HEAD_DIM;
  v16h qf[4];
#pragma unroll
  for (int kf = 0; kf < 4; ++kf)
    qf[kf] = load_a_frag(Qbase + kf * 32, HEAD_DIM, lane);

  v8f o[8];
#pragma unroll
  for (int i = 0; i < 8; ++i) o[i] = zero_v8f();
  float m_i[8], l_i[8];
#pragma unroll
  for (int r = 0; r < 8; ++r) { m_i[r] = -1e30f; l_i[r] = 0.f; }

  half_t* myP = &Pl[w][0];
  const int nblocks = (qs + 16 + 31) >> 5;   // causal: keys 0 .. qs+15

  unsigned ldsK[2], ldsV[2];
#pragma unroll
  for (int s = 0; s < 2; ++s) {
    ldsK[s] = (unsigned)(unsigned long long)(&Ksh[s][0]);
    ldsV[s] = (unsigned)(unsigned long long)(&Vsh[s][0]);
  }

  // prologue: TDM stage 0 (K tile 32x128, V tile 128x32), wave 0 only
  if (w == 0) {
    tdm_load_2d(ldsK[0], Khead, /*tile*/HEAD_DIM, 32,
                /*tensor*/HEAD_DIM, SEQ, /*stride0*/HEAD_DIM, 5, 3);
    tdm_load_2d(ldsV[0], Vhead, /*tile*/32, HEAD_DIM,
                /*tensor*/SEQ, HEAD_DIM, /*stride0*/SEQ, 3, 3);
  }

  for (int kb = 0; kb < nblocks; ++kb) {
    const int buf = kb & 1;
    if (w == 0) {
      if (kb + 1 < nblocks) {
        const int k1 = (kb + 1) * 32;
        tdm_load_2d(ldsK[buf ^ 1], Khead + (size_t)k1 * HEAD_DIM, HEAD_DIM, 32,
                    HEAD_DIM, SEQ, HEAD_DIM, 5, 3);
        tdm_load_2d(ldsV[buf ^ 1], Vhead + k1, 32, HEAD_DIM,
                    SEQ, HEAD_DIM, SEQ, 3, 3);
        __builtin_amdgcn_s_wait_tensorcnt((short)2);
      } else {
        __builtin_amdgcn_s_wait_tensorcnt((short)0);
      }
    }
    __syncthreads();   // stage kb tiles visible to all waves

    // ---- scores: Q(16x128) @ K^T(128x32) -> two 16x16 C frags (from LDS K tile)
    v8f sc0 = zero_v8f(), sc1 = zero_v8f();
#pragma unroll
    for (int kf = 0; kf < 4; ++kf) {
      v16h kb0 = load_b_frag(&Ksh[buf][0]        + kf * 32, KLD, lane);
      v16h kb1 = load_b_frag(&Ksh[buf][16 * KLD] + kf * 32, KLD, lane);
      sc0 = wmma_f16(qf[kf], kb0, sc0);
      sc1 = wmma_f16(qf[kf], kb1, sc1);
    }

    // ---- online softmax (per-row stats live in lanes of the same g-group)
    const int k0 = kb * 32;
    const int key0 = k0 + ln, key1 = key0 + 16;
    float alpha[8];
#pragma unroll
    for (int r = 0; r < 8; ++r) {
      const int q = qs + r + 8 * g;
      float s0 = (key0 <= q) ? sc0[r] : -1e30f;
      float s1 = (key1 <= q) ? sc1[r] : -1e30f;
      float mx = fmaxf(s0, s1);
#pragma unroll
      for (int off = 8; off; off >>= 1) mx = fmaxf(mx, __shfl_xor(mx, off, 32));
      const float mN = fmaxf(m_i[r], mx);
      const float a  = __expf(m_i[r] - mN);
      const float p0 = __expf(s0 - mN);
      const float p1 = __expf(s1 - mN);
      float ps = p0 + p1;
#pragma unroll
      for (int off = 8; off; off >>= 1) ps += __shfl_xor(ps, off, 32);
      l_i[r] = l_i[r] * a + ps;
      m_i[r] = mN;
      alpha[r] = a;
      myP[(r + 8 * g) * PLD + ln]      = (half_t)p0;
      myP[(r + 8 * g) * PLD + 16 + ln] = (half_t)p1;
    }

#pragma unroll
    for (int dt = 0; dt < 8; ++dt)
#pragma unroll
      for (int r = 0; r < 8; ++r) o[dt][r] *= alpha[r];

    // wave-local LDS RAW: make the P stores visible before fragment reload
    asm volatile("s_wait_dscnt 0x0" ::: "memory");

    // ---- PV: P(16x32) @ V(32x128) from LDS V tile
    v16h pf = load_a_frag(myP, PLD, lane);
#pragma unroll
    for (int dt = 0; dt < 8; ++dt) {
      v16h vf = load_b_frag(&Vsh[buf][dt * 16 * VLD], VLD, lane);
      o[dt] = wmma_f16(pf, vf, o[dt]);
    }

    __syncthreads();   // all waves done with buf; next TDM stage may overwrite it
  }

  // ---- normalize and store ctx as (B*S, D_MODEL) f16 for the O projection
#pragma unroll
  for (int r = 0; r < 8; ++r) {
    const float inv = 1.f / l_i[r];
    const size_t base =
        (size_t)(b * SEQ + qs + r + 8 * g) * D_MODEL + (size_t)h * HEAD_DIM;
#pragma unroll
    for (int dt = 0; dt < 8; ++dt)
      ctx[base + dt * 16 + ln] = (half_t)(o[dt][r] * inv);
  }
}

// ---------------------------------------------------------------- launch

extern "C" void kernel_launch(void* const* d_in, const int* in_sizes, int n_in,
                              void* d_out, int out_size, void* d_ws, size_t ws_size,
                              hipStream_t stream) {
  const float* x  = (const float*)d_in[0];
  // d_in[1] = causal_mask (unused; causality handled analytically)
  const float* Wq = (const float*)d_in[2];
  const float* bq = (const float*)d_in[3];
  const float* Wk = (const float*)d_in[4];
  const float* bk = (const float*)d_in[5];
  const float* Wv = (const float*)d_in[6];
  const float* bv = (const float*)d_in[7];
  const float* Wo = (const float*)d_in[8];
  const float* bo = (const float*)d_in[9];
  float* out = (float*)d_out;

  char* ws = (char*)d_ws;
  size_t off = 0;
  auto alloc = [&](size_t elems) {
    void* p = ws + off;
    off += elems * sizeof(half_t);
    return (half_t*)p;
  };
  const size_t NW = (size_t)D_MODEL * D_MODEL;
  const size_t NX = (size_t)MTOT * D_MODEL;
  half_t* xh   = alloc(NX);
  half_t* Wqh  = alloc(NW);
  half_t* Wkh  = alloc(NW);
  half_t* Wvh  = alloc(NW);
  half_t* Woh  = alloc(NW);
  half_t* Qh   = alloc(NX);
  half_t* Khb  = alloc(NX);
  half_t* Vth  = alloc(NX);
  half_t* ctxh = alloc(NX);

  auto cvt = [&](const float* s, half_t* d, size_t n) {
    int n4 = (int)(n / 4);
    cvt_f32_f16<<<(n4 + 255) / 256, 256, 0, stream>>>(s, d, n4);
  };
  cvt(x,  xh,  NX);
  cvt(Wq, Wqh, NW);
  cvt(Wk, Wkh, NW);
  cvt(Wv, Wvh, NW);
  cvt(Wo, Woh, NW);

  dim3 gg(D_MODEL / 128, MTOT / 128);  // (16, 32)
  gemm_wmma<<<gg, 256, 0, stream>>>(xh, Wqh, bq, Qh,  nullptr, 0);
  gemm_wmma<<<gg, 256, 0, stream>>>(xh, Wkh, bk, Khb, nullptr, 1);
  gemm_wmma<<<gg, 256, 0, stream>>>(xh, Wvh, bv, Vth, nullptr, 2);

  dim3 fg(SEQ / 64, NUM_HEADS, BATCH);
  flash_attn<<<fg, 128, 0, stream>>>(Qh, Khb, Vth, ctxh);

  gemm_wmma<<<gg, 256, 0, stream>>>(ctxh, Woh, bo, nullptr, out, 3);
}
